// MultiHeadAttention_3384434230188
// MI455X (gfx1250) — matrix operand order repaired and checked by reading, not compiled, not run on hardware
//
#include <hip/hip_runtime.h>
#include <hip/hip_bf16.h>

typedef __attribute__((ext_vector_type(16))) __bf16   v16bf;
typedef __attribute__((ext_vector_type(8)))  float    v8f;
typedef __attribute__((ext_vector_type(4)))  unsigned v4u;
typedef __attribute__((ext_vector_type(8)))  int      v8i;
typedef __attribute__((ext_vector_type(4)))  int      v4i;

#define DDIM 1024
#define SEQ  2048
#define NH   16
#define DH   64
#define BB   2
#define MTOT (BB * SEQ)   // 4096 rows for all projection GEMMs

// ---- attention LDS layout (dynamic LDS block; base offset 0, no static LDS) ----
#define SC_BYTES   (16 * SEQ * 4)            // 131072  f32 scores [16][2048]
#define PB_BYTES   (16 * SEQ * 2)            // 65536   bf16 probs [16][2048]
#define RED_BYTES  ((16 * 8 + 16) * 4)       // 576     reduce buffers
#define VBUF_OFF   (SC_BYTES + PB_BYTES + RED_BYTES)   // 197184 (16B aligned)
#define VROW_ELEMS 264                        // 256 data + 8 pad bf16 (TDM pad)
#define VBUF_ELEMS (64 * VROW_ELEMS)          // 16896 bf16 = 33792 B per buffer
#define VBUF_BYTES (VBUF_ELEMS * 2)
#define SMEM_TOTAL (VBUF_OFF + 2 * VBUF_BYTES)  // 264768 B < 320 KB

__device__ __forceinline__ __bf16 f2bf(float f) { return (__bf16)f; }

// -----------------------------------------------------------------------------
// TDM: DMA a [64 rows x 256 elem] bf16 tile (row stride 2048 elem) from global
// into LDS at byte offset lds_off, padding 4 dwords after every 128 dwords so
// the LDS row stride becomes 132 dwords (bank-conflict-free strided reads).
// D# per CDNA5 ISA ch.8: group0 = {count/lds_addr/global_addr/type},
// group1 = {flags+pad, tensor dims, tile dims, strides}.
// clang-23 lane builtin: (uint32x4, int32x8, int32x4, int32x4, int32x8, i32).
// -----------------------------------------------------------------------------
__device__ __forceinline__ void tdm_load_vtile(const __bf16* gsrc, unsigned lds_off)
{
    const unsigned long long ga = (unsigned long long)(const void*)gsrc;
    v4u g0 = { 1u,                                   // count=1, user descriptor
               lds_off,                              // lds_addr [63:32]
               (unsigned)ga,                         // global_addr [95:64]
               (unsigned)((ga >> 32) & 0x1FFFFFFull) // global_addr [120:96]
                   | (2u << 30) };                   // type=2 ("image")
    v8i g1 = { (int)((1u << 16)      // data_size = 1 -> 2-byte elements
                   | (1u << 20)      // pad_enable
                   | (6u << 22)      // pad_interval: code 6 = 128 dwords (512 B)
                   | (3u << 25)),    // pad_amount:  code 3 = 4 dwords (16 B)
               (int)(256u << 16),    // tensor_dim0 = 256 (low16 at [63:48])
               (int)(64u  << 16),    // tensor_dim1 = 64  (low16 at [95:80])
               (int)(256u << 16),    // tile_dim0   = 256 ([127:112])
               64,                   // tile_dim1   = 64  ([143:128])
               2048,                 // tensor_dim0_stride low32 ([191:160])
               0, 0 };
    v4i z4 = { 0, 0, 0, 0 };         // groups 2/3 unused (2-D tensor)
    v8i z8 = { 0, 0, 0, 0, 0, 0, 0, 0 };
    __builtin_amdgcn_tensor_load_to_lds(g0, g1, z4, z4, z8, 0);
}

// -----------------------------------------------------------------------------
// GEMM: out = A(4096 x 1024) @ W^T + bias     (torch Linear semantics)
//   MODE 0: bf16 out, head-split      [B, H, S, DH]
//   MODE 1: bf16 out, head-transposed [B, H, DH, S]
//   MODE 2: f32  out, row-major       [4096, 1024]
// One wave computes a 16x64 tile (A reused across 4 WMMAs); 4 waves / block.
// -----------------------------------------------------------------------------
template <int MODE, bool ABF16>
__global__ void gemm_proj(const void* __restrict__ Ain,
                          const float* __restrict__ W,
                          const float* __restrict__ bias,
                          void* __restrict__ outp)
{
    const int wave = threadIdx.x >> 5;
    const int lane = threadIdx.x & 31;
    const int row  = lane & 15;
    const int half = lane >> 4;
    const int m0 = blockIdx.x * 64 + wave * 16;
    const int n0 = blockIdx.y * 64;

    v8f acc[4] = {};

    for (int k0 = 0; k0 < DDIM; k0 += 32) {
        v16bf a;
        if (ABF16) {
            const __bf16* Arow = (const __bf16*)Ain + (size_t)(m0 + row) * DDIM + k0;
            if (k0 + 32 < DDIM) __builtin_prefetch(Arow + 32, 0, 3);
#pragma unroll
            for (int vv = 0; vv < 8; ++vv) {
                const int kk = (half << 4) + (vv << 1);
                a[2 * vv]     = Arow[kk];
                a[2 * vv + 1] = Arow[kk + 1];
            }
        } else {
            const float* Arow = (const float*)Ain + (size_t)(m0 + row) * DDIM + k0;
            if (k0 + 32 < DDIM) __builtin_prefetch(Arow + 32, 0, 3);
#pragma unroll
            for (int vv = 0; vv < 8; ++vv) {
                const int kk = (half << 4) + (vv << 1);
                a[2 * vv]     = f2bf(Arow[kk]);
                a[2 * vv + 1] = f2bf(Arow[kk + 1]);
            }
        }
#pragma unroll
        for (int t = 0; t < 4; ++t) {
            const int n = n0 + t * 16 + row;
            const float* Wrow = W + (size_t)n * DDIM + k0 + (half << 4);
            v16bf bm;
#pragma unroll
            for (int vv = 0; vv < 8; ++vv) {
                bm[2 * vv]     = f2bf(Wrow[2 * vv]);
                bm[2 * vv + 1] = f2bf(Wrow[2 * vv + 1]);
            }
            acc[t] = __builtin_amdgcn_wmma_f32_16x16x32_bf16(
                false, a, false, bm, (short)0, acc[t], false, false);
        }
    }

#pragma unroll
    for (int t = 0; t < 4; ++t) {
#pragma unroll
        for (int r = 0; r < 8; ++r) {
            const int M = m0 + r + (half << 3);
            const int n = n0 + t * 16 + row;
            const float val = acc[t][r] + bias[n];
            if (MODE == 2) {
                ((float*)outp)[(size_t)M * DDIM + n] = val;
            } else {
                const int b = M >> 11, s = M & (SEQ - 1);
                const int h = n >> 6,  d = n & (DH - 1);
                if (MODE == 0)
                    ((__bf16*)outp)[(((size_t)(b * NH + h)) * SEQ + s) * DH + d] = f2bf(val);
                else
                    ((__bf16*)outp)[(((size_t)(b * NH + h)) * DH + d) * SEQ + s] = f2bf(val);
            }
        }
    }
}

// -----------------------------------------------------------------------------
// Attention: one block = (b, h, 16-query strip). blockDim = 128 (4 waves).
// Phase 1: QK^T -> scaled scores in LDS (waves split the 128 k-tiles).
// Phase 2: cooperative softmax; fp32 probs -> global `a`, bf16 probs -> LDS.
// Phase 3: P @ V with TDM double-buffered V staging in LDS.
// -----------------------------------------------------------------------------
__global__ void attention_kernel(const __bf16* __restrict__ qh,
                                 const __bf16* __restrict__ kh,
                                 const __bf16* __restrict__ vt,
                                 float* __restrict__ a_out,
                                 __bf16* __restrict__ o_pre)
{
    extern __shared__ char smem[];
    float*  sc    = (float*)smem;                         // [16][SEQ] f32
    __bf16* pb    = (__bf16*)(smem + SC_BYTES);           // [16][SEQ] bf16
    float*  red   = (float*)(smem + SC_BYTES + PB_BYTES); // [16][8]
    float*  rstat = red + 16 * 8;                         // [16]
    __bf16* vbuf  = (__bf16*)(smem + VBUF_OFF);           // 2 x [64][264] bf16

    const int tid  = threadIdx.x;
    const int wave = tid >> 5;
    const int lane = tid & 31;
    const int row  = lane & 15;
    const int half = lane >> 4;

    const int b  = blockIdx.z, h = blockIdx.y;
    const int q0 = blockIdx.x * 16;
    const size_t bh = (size_t)(b * NH + h);

    const __bf16* Qb = qh + bh * SEQ * DH;   // [S][DH]
    const __bf16* Kb = kh + bh * SEQ * DH;   // [S][DH]
    const __bf16* Vb = vt + bh * DH * SEQ;   // [DH][S] (transposed)

    // ---- Phase 1: scores (Q tile hoisted, invariant across k-tiles)
    v16bf aq[2];
#pragma unroll
    for (int kc = 0; kc < 2; ++kc) {
        const __bf16* Arow = Qb + (size_t)(q0 + row) * DH + kc * 32;
#pragma unroll
        for (int vv = 0; vv < 8; ++vv) {
            const int kk = (half << 4) + (vv << 1);
            aq[kc][2 * vv]     = Arow[kk];
            aq[kc][2 * vv + 1] = Arow[kk + 1];
        }
    }
    for (int kt = wave * 32; kt < wave * 32 + 32; ++kt) {
        const int kp0 = kt * 16;
        v8f c = {};
#pragma unroll
        for (int kc = 0; kc < 2; ++kc) {
            const __bf16* Krow = Kb + (size_t)(kp0 + row) * DH + kc * 32 + (half << 4);
            v16bf bm;
#pragma unroll
            for (int vv = 0; vv < 8; ++vv) {
                bm[2 * vv]     = Krow[2 * vv];
                bm[2 * vv + 1] = Krow[2 * vv + 1];
            }
            c = __builtin_amdgcn_wmma_f32_16x16x32_bf16(
                false, aq[kc], false, bm, (short)0, c, false, false);
        }
#pragma unroll
        for (int r = 0; r < 8; ++r)
            sc[(size_t)(r + (half << 3)) * SEQ + kp0 + row] = c[r] * 0.125f;
    }
    __syncthreads();

    // ---- Phase 2: softmax (8 threads per row, 256-wide segments)
    const int r   = tid & 15;
    const int seg = tid >> 4;
    const int c0  = seg * (SEQ / 8);
    float m = -3.4e38f;
    for (int cc = 0; cc < SEQ / 8; ++cc)
        m = fmaxf(m, sc[(size_t)r * SEQ + c0 + cc]);
    red[r * 8 + seg] = m;
    __syncthreads();
    if (tid < 16) {
        float mm = red[tid * 8];
        for (int i2 = 1; i2 < 8; ++i2) mm = fmaxf(mm, red[tid * 8 + i2]);
        rstat[tid] = mm;
    }
    __syncthreads();
    const float rowmax = rstat[r];
    float ssum = 0.f;
    for (int cc = 0; cc < SEQ / 8; ++cc) {
        const float e = __expf(sc[(size_t)r * SEQ + c0 + cc] - rowmax);
        sc[(size_t)r * SEQ + c0 + cc] = e;
        ssum += e;
    }
    red[r * 8 + seg] = ssum;
    __syncthreads();
    if (tid < 16) {
        float s2 = 0.f;
        for (int i2 = 0; i2 < 8; ++i2) s2 += red[tid * 8 + i2];
        rstat[tid] = 1.0f / s2;
    }
    __syncthreads();
    const float inv = rstat[r];
    float* arow = a_out + (bh * SEQ + q0 + r) * (size_t)SEQ;
    for (int cc = 0; cc < SEQ / 8; ++cc) {
        const float p = sc[(size_t)r * SEQ + c0 + cc] * inv;
        arow[c0 + cc] = p;
        pb[(size_t)r * SEQ + c0 + cc] = f2bf(p);
    }
    __syncthreads();

    // ---- Phase 3: o = P @ V, V staged by TDM (double buffered, 8 x 256-kpos chunks)
    const int dh0 = wave * 16;
    v8f c = {};

    if (wave == 0)                               // TDM issue is wave-level (EXEC ignored)
        tdm_load_vtile(Vb, VBUF_OFF);            // chunk 0 -> buffer 0

    for (int mc = 0; mc < 8; ++mc) {
        if (wave == 0) {
            if (mc + 1 < 8) {
                tdm_load_vtile(Vb + (mc + 1) * 256,
                               VBUF_OFF + (unsigned)(((mc + 1) & 1) * VBUF_BYTES));
                __builtin_amdgcn_s_wait_tensorcnt(1);  // chunk mc landed
            } else {
                __builtin_amdgcn_s_wait_tensorcnt(0);
            }
        }
        __syncthreads();                               // release consumers

        const __bf16* vb = vbuf + (size_t)(mc & 1) * VBUF_ELEMS;
#pragma unroll
        for (int kl = 0; kl < 8; ++kl) {               // 8 kchunks of 32 inside chunk
            const int kc = mc * 8 + kl;
            v16bf a, bm;
#pragma unroll
            for (int vv = 0; vv < 8; ++vv) {
                const int kk = kc * 32 + (half << 4) + (vv << 1);
                a[2 * vv]     = pb[(size_t)row * SEQ + kk];
                a[2 * vv + 1] = pb[(size_t)row * SEQ + kk + 1];
            }
            // B'(K=kpos, n=dh) from padded LDS tile: row stride 264 elems
            const __bf16* Vrow = vb + (size_t)(dh0 + row) * VROW_ELEMS
                                    + kl * 32 + (half << 4);
#pragma unroll
            for (int vv = 0; vv < 8; ++vv) {
                bm[2 * vv]     = Vrow[2 * vv];
                bm[2 * vv + 1] = Vrow[2 * vv + 1];
            }
            c = __builtin_amdgcn_wmma_f32_16x16x32_bf16(
                false, a, false, bm, (short)0, c, false, false);
        }
        __syncthreads();                               // buffer consumed; safe to refill
    }

#pragma unroll
    for (int rr = 0; rr < 8; ++rr) {
        const int M = q0 + rr + (half << 3);
        o_pre[((size_t)(b * SEQ) + M) * DDIM + h * DH + dh0 + row] = f2bf(c[rr]);
    }
}

// -----------------------------------------------------------------------------
extern "C" void kernel_launch(void* const* d_in, const int* in_sizes, int n_in,
                              void* d_out, int out_size, void* d_ws, size_t ws_size,
                              hipStream_t stream)
{
    (void)in_sizes; (void)n_in; (void)out_size; (void)ws_size;

    const float* q  = (const float*)d_in[0];
    const float* k  = (const float*)d_in[1];
    const float* v  = (const float*)d_in[2];
    const float* Wq = (const float*)d_in[3];
    const float* bq = (const float*)d_in[4];
    const float* Wk = (const float*)d_in[5];
    const float* bk = (const float*)d_in[6];
    const float* Wv = (const float*)d_in[7];
    const float* bv = (const float*)d_in[8];
    const float* Wo = (const float*)d_in[9];
    const float* bo = (const float*)d_in[10];

    float* o_out = (float*)d_out;                       // [B,S,D]
    float* a_out = o_out + (size_t)BB * SEQ * DDIM;     // [B,H,S,S]

    char* w = (char*)d_ws;
    const size_t HBYTES = (size_t)BB * NH * SEQ * DH * sizeof(__bf16); // 8 MB each
    __bf16* qh    = (__bf16*)(w);                 // [B,H,S,DH]
    __bf16* kh    = (__bf16*)(w + HBYTES);        // [B,H,S,DH]
    __bf16* vt    = (__bf16*)(w + 2 * HBYTES);    // [B,H,DH,S]
    __bf16* o_pre = (__bf16*)(w + 3 * HBYTES);    // [B*S, D]

    const dim3 ggrid(MTOT / 64, DDIM / 64);       // 64 x 16 blocks, 128 thr (4 waves)
    gemm_proj<0, false><<<ggrid, 128, 0, stream>>>(q, Wq, bq, qh);
    gemm_proj<0, false><<<ggrid, 128, 0, stream>>>(k, Wk, bk, kh);
    gemm_proj<1, false><<<ggrid, 128, 0, stream>>>(v, Wv, bv, vt);

    const dim3 agrid(SEQ / 16, NH, BB);           // 128 x 16 x 2 blocks
    attention_kernel<<<agrid, 128, SMEM_TOTAL, stream>>>(qh, kh, vt, a_out, o_pre);

    gemm_proj<2, true><<<ggrid, 128, 0, stream>>>(o_pre, Wo, bo, o_out);
}
